// ChunkedKasminaLayerV2_22883585753481
// MI455X (gfx1250) — matrix operand order repaired and checked by reading, not compiled, not run on hardware
//
#include <hip/hip_runtime.h>

#define B_SZ   8192
#define D_INN  4096
#define D_OUTT 4096
#define S_SEED 64
#define C_CH   64
#define H_DIM  256

typedef __attribute__((ext_vector_type(16))) _Float16 h16v;
typedef __attribute__((ext_vector_type(4)))  _Float16 h4v;
typedef __attribute__((ext_vector_type(8)))  float    f8v;

union Frag16 { h16v v; uint4 q[2]; };

typedef _Float16 Row72[72];

__global__ __launch_bounds__(256)
void kasmina_fused(const float* __restrict__ x,
                   const int*   __restrict__ states,
                   const float* __restrict__ alphas,
                   const float* __restrict__ base_w,
                   const float* __restrict__ base_b,
                   const float* __restrict__ w1,
                   const float* __restrict__ b1,
                   const float* __restrict__ w2,
                   const float* __restrict__ b2,
                   float* __restrict__ out)
{
    // 54 KB arena, aliased between stage-1 double buffers and stage-2 buffers.
    __shared__ __align__(16) unsigned char smem[55296];
    Row72* Abuf0 = (Row72*)(smem);           // 128 rows f16: stage-1 x tile, buf 0
    Row72* Abuf1 = (Row72*)(smem + 18432);   // 128 rows f16: stage-1 x tile, buf 1
    Row72* Bbuf0 = (Row72*)(smem + 36864);   //  64 rows f16: stage-1 w tile, buf 0
    Row72* Bbuf1 = (Row72*)(smem + 46080);   //  64 rows f16: stage-1 w tile, buf 1
    // stage-2 aliases (all reuse separated by barriers):
    Row72* CH  = (Row72*)(smem);             // chunk f16 (A of GEMM1), 128 rows
    Row72* HL  = (Row72*)(smem + 18432);     // relu(h) slice (A of GEMM2), 128 rows
    Row72* W1s = (Row72*)(smem + 36864);     // w1 slice [h][c], 64 rows
    Row72* W2s = (Row72*)(smem + 46080);     // w2 slice [c][h], 64 rows

    const int tid  = threadIdx.x;
    const int wave = tid >> 5;
    const int lane = tid & 31;
    const int l16  = lane & 15;
    const int lh   = lane >> 4;

    const int row0 = blockIdx.x * 128;   // batch rows
    const int s    = blockIdx.y;         // seed

    f8v acc[4];
    #pragma unroll
    for (int nt = 0; nt < 4; ++nt)
        #pragma unroll
        for (int v = 0; v < 8; ++v) acc[nt][v] = 0.0f;

    // in-flight registers for the next K tile (128x64 A + 64x64 B, as float4)
    float4 areg[8], breg[4];

    auto load_tile = [&](int k0) {
        #pragma unroll
        for (int i = 0; i < 8; ++i) {
            const int idx = tid + i * 256;   // 2048 float4s: A tile
            const int r   = idx >> 4;
            const int kq  = idx & 15;
            areg[i] = *(const float4*)(x + (size_t)(row0 + r) * D_INN + k0 + kq * 4);
        }
        #pragma unroll
        for (int i = 0; i < 4; ++i) {
            const int idx = tid + i * 256;   // 1024 float4s: B tile
            const int n   = idx >> 4;
            const int kq  = idx & 15;
            breg[i] = *(const float4*)(base_w + (size_t)(s * 64 + n) * D_INN + k0 + kq * 4);
        }
    };
    auto store_tile = [&](int buf) {
        Row72* Ad = buf ? Abuf1 : Abuf0;
        Row72* Bd = buf ? Bbuf1 : Bbuf0;
        #pragma unroll
        for (int i = 0; i < 8; ++i) {
            const int idx = tid + i * 256;
            const int r   = idx >> 4;
            const int kq  = idx & 15;
            h4v hv = { (_Float16)areg[i].x, (_Float16)areg[i].y,
                       (_Float16)areg[i].z, (_Float16)areg[i].w };
            *(h4v*)(&Ad[r][kq * 4]) = hv;
        }
        #pragma unroll
        for (int i = 0; i < 4; ++i) {
            const int idx = tid + i * 256;
            const int n   = idx >> 4;
            const int kq  = idx & 15;
            h4v hv = { (_Float16)breg[i].x, (_Float16)breg[i].y,
                       (_Float16)breg[i].z, (_Float16)breg[i].w };
            *(h4v*)(&Bd[n][kq * 4]) = hv;
        }
    };

    // ---------------- Stage 1: chunk = x_tile @ base_w[s*64:(s+1)*64, :].T ----------------
    const int NT = D_INN / 64;
    load_tile(0);
    store_tile(0);
    __syncthreads();

    for (int kt = 0; kt < NT; ++kt) {
        const int cur = kt & 1;
        if (kt + 1 < NT) load_tile((kt + 1) * 64);   // global loads overlap compute below

        Row72* Ab = cur ? Abuf1 : Abuf0;
        Row72* Bt = cur ? Bbuf1 : Bbuf0;
        const int m = wave * 16 + l16;
        #pragma unroll
        for (int kh = 0; kh < 2; ++kh) {
            Frag16 a;
            const int kb = kh * 32 + lh * 8;
            a.q[0] = *(const uint4*)(&Ab[m][kb]);
            a.q[1] = *(const uint4*)(&Ab[m][kb + 16]);
            Frag16 b[4];
            const int bk = kh * 32 + lh * 8;
            #pragma unroll
            for (int nt = 0; nt < 4; ++nt) {
                b[nt].q[0] = *(const uint4*)(&Bt[nt * 16 + l16][bk]);
                b[nt].q[1] = *(const uint4*)(&Bt[nt * 16 + l16][bk + 16]);
            }
            #pragma unroll
            for (int nt = 0; nt < 4; ++nt)
                acc[nt] = __builtin_amdgcn_wmma_f32_16x16x32_f16(
                    false, a.v, false, b[nt].v, (short)0, acc[nt], false, false);
        }

        // Write-target buffer's last readers finished at the previous end-of-iter barrier.
        if (kt + 1 < NT) store_tile(1 - cur);
        __syncthreads();
    }

    // bias + stash chunk (f16) into CH as stage-2 A matrix [batch][c]
    #pragma unroll
    for (int nt = 0; nt < 4; ++nt) {
        const float bbn = base_b[s * 64 + nt * 16 + l16];
        #pragma unroll
        for (int v = 0; v < 8; ++v) {
            acc[nt][v] += bbn;
            CH[wave * 16 + v + lh * 8][nt * 16 + l16] = (_Float16)acc[nt][v];
        }
    }

    // bp accumulators, init with b2
    f8v bp[4];
    #pragma unroll
    for (int nt = 0; nt < 4; ++nt) {
        const float b2n = b2[s * 64 + nt * 16 + l16];
        #pragma unroll
        for (int v = 0; v < 8; ++v) bp[nt][v] = b2n;
    }

    // ---------------- Stage 2: seed MLP, H processed in 4 slices of 64 ----------------
    for (int n0 = 0; n0 < H_DIM; n0 += 64) {
        __syncthreads();   // previous slice done reading W1s/W2s
        // W1 slice: W1s[h][c] = w1[s][c][n0+h]  (K=c contiguous per output column h)
        #pragma unroll
        for (int i = 0; i < 4; ++i) {
            const int idx = tid + i * 256;
            const int c   = idx >> 4;
            const int hq  = idx & 15;
            const float4 vv = *(const float4*)(w1 + (size_t)(s * 64 + c) * H_DIM + n0 + hq * 4);
            W1s[hq * 4 + 0][c] = (_Float16)vv.x;
            W1s[hq * 4 + 1][c] = (_Float16)vv.y;
            W1s[hq * 4 + 2][c] = (_Float16)vv.z;
            W1s[hq * 4 + 3][c] = (_Float16)vv.w;
        }
        // W2 slice: W2s[c][h] = w2[s][n0+h][c]  (K=h contiguous per output column c)
        #pragma unroll
        for (int i = 0; i < 4; ++i) {
            const int idx = tid + i * 256;
            const int hr  = idx >> 4;
            const int cq  = idx & 15;
            const float4 vv = *(const float4*)(w2 + (size_t)(s * H_DIM + n0 + hr) * C_CH + cq * 4);
            W2s[cq * 4 + 0][hr] = (_Float16)vv.x;
            W2s[cq * 4 + 1][hr] = (_Float16)vv.y;
            W2s[cq * 4 + 2][hr] = (_Float16)vv.z;
            W2s[cq * 4 + 3][hr] = (_Float16)vv.w;
        }
        __syncthreads();

        // GEMM1: h_slice = relu(chunk @ w1_slice + b1_slice), K = 64 (c)
        f8v hacc[4];
        #pragma unroll
        for (int nt = 0; nt < 4; ++nt) {
            const float b1n = b1[s * H_DIM + n0 + nt * 16 + l16];
            #pragma unroll
            for (int v = 0; v < 8; ++v) hacc[nt][v] = b1n;
        }
        const int m = wave * 16 + l16;
        #pragma unroll
        for (int kh = 0; kh < 2; ++kh) {
            Frag16 a;
            const int kb = kh * 32 + lh * 8;
            a.q[0] = *(const uint4*)(&CH[m][kb]);
            a.q[1] = *(const uint4*)(&CH[m][kb + 16]);
            Frag16 b[4];
            const int bk = kh * 32 + lh * 8;
            #pragma unroll
            for (int nt = 0; nt < 4; ++nt) {
                b[nt].q[0] = *(const uint4*)(&W1s[nt * 16 + l16][bk]);
                b[nt].q[1] = *(const uint4*)(&W1s[nt * 16 + l16][bk + 16]);
            }
            #pragma unroll
            for (int nt = 0; nt < 4; ++nt)
                hacc[nt] = __builtin_amdgcn_wmma_f32_16x16x32_f16(
                    false, a.v, false, b[nt].v, (short)0, hacc[nt], false, false);
        }
        // relu + stash into HL (wave-private rows: no workgroup barrier needed)
        #pragma unroll
        for (int nt = 0; nt < 4; ++nt)
            #pragma unroll
            for (int v = 0; v < 8; ++v) {
                float hv = hacc[nt][v];
                hv = hv > 0.0f ? hv : 0.0f;
                HL[wave * 16 + v + lh * 8][nt * 16 + l16] = (_Float16)hv;
            }

        // GEMM2: bp += h_slice @ w2_slice, K = 64 (h)
        #pragma unroll
        for (int kh = 0; kh < 2; ++kh) {
            Frag16 a;
            const int kb = kh * 32 + lh * 8;
            a.q[0] = *(const uint4*)(&HL[m][kb]);
            a.q[1] = *(const uint4*)(&HL[m][kb + 16]);
            Frag16 b[4];
            const int bk = kh * 32 + lh * 8;
            #pragma unroll
            for (int nt = 0; nt < 4; ++nt) {
                b[nt].q[0] = *(const uint4*)(&W2s[nt * 16 + l16][bk]);
                b[nt].q[1] = *(const uint4*)(&W2s[nt * 16 + l16][bk + 16]);
            }
            #pragma unroll
            for (int nt = 0; nt < 4; ++nt)
                bp[nt] = __builtin_amdgcn_wmma_f32_16x16x32_f16(
                    false, a.v, false, b[nt].v, (short)0, bp[nt], false, false);
        }
    }

    // ---------------- Blend + store ----------------
    // bp_sg == bp in forward => out = c_base*chunk + (c_bp + c_sg)*bp
    const int   st = states[s];
    const float al = alphas[s];
    const float c_base = (st == 2) ? (1.0f - al) : (st == 3) ? 0.1f : (st == 4) ? 0.0f : 1.0f;
    const float c_tot  = (st == 2) ? al          : (st == 3) ? 0.9f : (st == 4) ? 1.0f : 0.0f;

    #pragma unroll
    for (int nt = 0; nt < 4; ++nt)
        #pragma unroll
        for (int v = 0; v < 8; ++v) {
            const int row = row0 + wave * 16 + v + lh * 8;
            const int col = s * 64 + nt * 16 + l16;
            out[(size_t)row * D_OUTT + col] = c_base * acc[nt][v] + c_tot * bp[nt][v];
        }
}

extern "C" void kernel_launch(void* const* d_in, const int* in_sizes, int n_in,
                              void* d_out, int out_size, void* d_ws, size_t ws_size,
                              hipStream_t stream) {
    const float* x      = (const float*)d_in[0];
    const int*   states = (const int*)  d_in[1];
    const float* alphas = (const float*)d_in[2];
    const float* base_w = (const float*)d_in[3];
    const float* base_b = (const float*)d_in[4];
    const float* w1     = (const float*)d_in[5];
    const float* b1     = (const float*)d_in[6];
    const float* w2     = (const float*)d_in[7];
    const float* b2     = (const float*)d_in[8];
    float* out = (float*)d_out;

    dim3 grid(B_SZ / 128, S_SEED);
    kasmina_fused<<<grid, 256, 0, stream>>>(x, states, alphas, base_w, base_b,
                                            w1, b1, w2, b2, out);
}
